// TimeThenSpaceModel_25778393710600
// MI455X (gfx1250) — compile-verified, matrix operand order checked by reading, not run on hardware
//
#include <hip/hip_runtime.h>
#include <hip/hip_bf16.h>
#include <math.h>

// Problem constants (from reference)
#define B_DIM 8
#define T_DIM 12
#define N_DIM 20000
#define F_INN 2
#define H_DIM 64
#define G3    192          // 3*H
#define HOR   12
#define E_DIM 320000
#define BN    (B_DIM * N_DIM)   // 160000
#define GRU_WAVES 4             // waves per GRU block sharing one weight copy

typedef __attribute__((ext_vector_type(16))) _Float16 v16h;
typedef __attribute__((ext_vector_type(8)))  float    v8f;

// K index inside a 16-bit WMMA A/B fragment (ISA 7.12.2):
// lanes 0-15 hold K {0..7,16..23}; lanes 16-31 hold K {8..15,24..31}
__device__ __forceinline__ int frag_k(int lane, int j) {
  const int hi = (lane >> 4) & 1;
  return (j < 8) ? (hi * 8 + j) : (16 + hi * 8 + (j - 8));
}

__device__ __forceinline__ float sigmoidf_(float v) {
  return 1.0f / (1.0f + __expf(-v));
}

// ---------------------------------------------------------------------------
// Edge-weight normalization: deg = segment_sum(w, idx); w *= 1/deg[idx]
// ---------------------------------------------------------------------------
__global__ __launch_bounds__(256)
void deg_kernel(const float* __restrict__ ew, const int* __restrict__ src,
                const int* __restrict__ tgt, float* __restrict__ degf,
                float* __restrict__ degb) {
  const int e = blockIdx.x * blockDim.x + threadIdx.x;
  if (e >= E_DIM) return;
  const float w = ew[e];
  atomicAdd(&degf[tgt[e]], w);   // forward: normalize by in-degree at tgt
  atomicAdd(&degb[src[e]], w);   // backward: normalize by in-degree at src
}

__global__ __launch_bounds__(256)
void norm_kernel(const float* __restrict__ ew, const int* __restrict__ src,
                 const int* __restrict__ tgt, const float* __restrict__ degf,
                 const float* __restrict__ degb, float* __restrict__ wf,
                 float* __restrict__ wb) {
  const int e = blockIdx.x * blockDim.x + threadIdx.x;
  if (e >= E_DIM) return;
  const float w  = ew[e];
  const float df = degf[tgt[e]];
  const float db = degb[src[e]];
  wf[e] = (df > 0.0f) ? w / df : 0.0f;
  wb[e] = (db > 0.0f) ? w / db : 0.0f;
}

// ---------------------------------------------------------------------------
// Fused encoder + per-node GRU over time.
// 128-thread blocks = 4 waves sharing ONE f16 copy of w_ih/w_hh in dynamic
// LDS (48 KB); each wave privately owns a 16-sequence tile with its own
// h/xg/hg regions (14 KB/wave). Total 104 KB LDS -> 3 blocks (12 waves) per
// 320 KB WGP instead of the previous 5 single-wave blocks.
// ---------------------------------------------------------------------------
__global__ __launch_bounds__(32 * GRU_WAVES)
void gru_encode_kernel(const float* __restrict__ x,
                       const float* __restrict__ enc_w,
                       const float* __restrict__ enc_b,
                       const float* __restrict__ node_emb,
                       const float* __restrict__ w_ih,
                       const float* __restrict__ w_hh,
                       const float* __restrict__ b_ih,
                       const float* __restrict__ b_hh,
                       float* __restrict__ h_out) {
  extern __shared__ _Float16 smem[];
  _Float16* s_wih = smem;                        // [G3*H]  B[k][g] = w_ih[g][k]
  _Float16* s_whh = s_wih + G3 * H_DIM;          // [G3*H]
  const int tid  = threadIdx.x;
  const int lane = tid & 31;
  const int wv   = tid >> 5;
  _Float16* s_pw = s_whh + G3 * H_DIM + wv * (16 * H_DIM + 2 * 16 * G3);
  _Float16* s_h  = s_pw;                         // [16*H]  hidden state (f16)
  _Float16* s_xg = s_h + 16 * H_DIM;             // [16*G3] input-side gates
  _Float16* s_hg = s_xg + 16 * G3;               // [16*G3] hidden-side gates

  const int m    = lane & 15;
  const int hi   = lane >> 4;
  const int row0 = (blockIdx.x * GRU_WAVES + wv) * 16;
  const int seq  = row0 + m;
  const int b    = seq / N_DIM;
  const int n    = seq % N_DIM;

  // cooperative weight staging (f32 -> f16), all 4 waves
  for (int i = tid; i < G3 * H_DIM; i += 32 * GRU_WAVES) {
    s_wih[i] = (_Float16)w_ih[i];
    s_whh[i] = (_Float16)w_hh[i];
  }
  for (int i = lane; i < 16 * H_DIM; i += 32) s_h[i] = (_Float16)0.0f;
  __syncthreads();

  for (int t = 0; t < T_DIM; ++t) {
    const long long xb = (((long long)b * T_DIM + t) * N_DIM + n) * F_INN;
    const float x0 = x[xb + 0];
    const float x1 = x[xb + 1];

    // A fragments: x-side (encoded on the fly) and h-side, K = 0..63
    v16h ax0, ax1, ah0, ah1;
#pragma unroll
    for (int j = 0; j < 16; ++j) {
      const int k0 = frag_k(lane, j);
      const int k1 = 32 + k0;
      const float e0 = x0 * enc_w[k0] + x1 * enc_w[H_DIM + k0] + enc_b[k0] +
                       node_emb[(long long)n * H_DIM + k0];
      const float e1 = x0 * enc_w[k1] + x1 * enc_w[H_DIM + k1] + enc_b[k1] +
                       node_emb[(long long)n * H_DIM + k1];
      ax0[j] = (_Float16)e0;
      ax1[j] = (_Float16)e1;
      ah0[j] = s_h[m * H_DIM + k0];
      ah1[j] = s_h[m * H_DIM + k1];
    }

    // gates = [x_emb | h] @ [w_ih^T | w_hh^T]; 12 N-tiles of 16 over 192 cols
    for (int g = 0; g < 12; ++g) {
      const int ncol = g * 16 + m;
      v16h b0, b1;
      v8f cx = {0,0,0,0,0,0,0,0};
      v8f ch = {0,0,0,0,0,0,0,0};
#pragma unroll
      for (int j = 0; j < 16; ++j) {
        const int k0 = frag_k(lane, j);
        b0[j] = s_wih[ncol * H_DIM + k0];
        b1[j] = s_wih[ncol * H_DIM + 32 + k0];
      }
      cx = __builtin_amdgcn_wmma_f32_16x16x32_f16(false, ax0, false, b0, (short)0, cx, false, false);
      cx = __builtin_amdgcn_wmma_f32_16x16x32_f16(false, ax1, false, b1, (short)0, cx, false, false);
#pragma unroll
      for (int j = 0; j < 16; ++j) {
        const int k0 = frag_k(lane, j);
        b0[j] = s_whh[ncol * H_DIM + k0];
        b1[j] = s_whh[ncol * H_DIM + 32 + k0];
      }
      ch = __builtin_amdgcn_wmma_f32_16x16x32_f16(false, ah0, false, b0, (short)0, ch, false, false);
      ch = __builtin_amdgcn_wmma_f32_16x16x32_f16(false, ah1, false, b1, (short)0, ch, false, false);

      const float bi = b_ih[ncol];
      const float bh = b_hh[ncol];
#pragma unroll
      for (int r = 0; r < 8; ++r) {       // C layout: VGPR r -> M = r + 8*hi
        const int mm = r + hi * 8;
        s_xg[mm * G3 + ncol] = (_Float16)(cx[r] + bi);
        s_hg[mm * G3 + ncol] = (_Float16)(ch[r] + bh);
      }
    }
    __syncthreads();

    // GRU gate nonlinearity (torch gate math) on VALU
    for (int idx = lane; idx < 16 * H_DIM; idx += 32) {
      const int rr = idx >> 6;
      const int hc = idx & 63;
      const float xr = (float)s_xg[rr * G3 + hc];
      const float xz = (float)s_xg[rr * G3 + 64 + hc];
      const float xn = (float)s_xg[rr * G3 + 128 + hc];
      const float hr = (float)s_hg[rr * G3 + hc];
      const float hz = (float)s_hg[rr * G3 + 64 + hc];
      const float hn = (float)s_hg[rr * G3 + 128 + hc];
      const float rg = sigmoidf_(xr + hr);
      const float zg = sigmoidf_(xz + hz);
      const float ng = tanhf(xn + rg * hn);
      const float hp = (float)s_h[rr * H_DIM + hc];
      const float hv = (1.0f - zg) * ng + zg * hp;
      s_h[rr * H_DIM + hc] = (_Float16)hv;
      if (t == T_DIM - 1)
        h_out[(long long)(row0 + rr) * H_DIM + hc] = hv;
    }
    __syncthreads();
  }
}

// ---------------------------------------------------------------------------
// Graph diffusion scatter: hout[b,tgt,:] += hin[b,src,:] * w[e]
// ---------------------------------------------------------------------------
__global__ __launch_bounds__(256)
void prop_kernel(const float* __restrict__ hin, const int* __restrict__ sidx,
                 const int* __restrict__ tidx, const float* __restrict__ w,
                 float* __restrict__ hout) {
  const long long gid = (long long)blockIdx.x * blockDim.x + threadIdx.x;
  if (gid >= (long long)E_DIM * B_DIM * H_DIM) return;
  const int hc = (int)(gid & 63);
  const long long eb = gid >> 6;
  const int e = (int)(eb % E_DIM);
  const int b = (int)(eb / E_DIM);
  const int s = sidx[e];
  const int t = tidx[e];
  const float msg = hin[((long long)b * N_DIM + s) * H_DIM + hc] * w[e];
  atomicAdd(&hout[((long long)b * N_DIM + t) * H_DIM + hc], msg);
}

// ---------------------------------------------------------------------------
// One K-chunk (64) of z = concat(outs) @ filt_w, accumulated into acc.
// chunk==0 initializes acc with filt_b. 256 threads = 8 waves x 16-row tiles.
// ---------------------------------------------------------------------------
__global__ __launch_bounds__(256)
void filt_kernel(const float* __restrict__ in, const float* __restrict__ filt_w,
                 const float* __restrict__ filt_b, float* __restrict__ acc,
                 int chunk) {
  __shared__ _Float16 s_b[H_DIM * H_DIM];   // 8 KB: B[k][n] = filt_w[chunk*64+k][n]
  const int tid  = threadIdx.x;
  const int lane = tid & 31;
  const int wv   = tid >> 5;
  const int m    = lane & 15;
  const int hi   = lane >> 4;

  for (int i = tid; i < H_DIM * H_DIM; i += 256) {
    const int k = i >> 6, nn = i & 63;
    s_b[i] = (_Float16)filt_w[(chunk * H_DIM + k) * H_DIM + nn];
  }
  __syncthreads();

  const int row0 = (blockIdx.x * 8 + wv) * 16;   // BN=160000 = 1250*128, exact
  const long long rbase = (long long)(row0 + m) * H_DIM;
  v16h a0, a1;
#pragma unroll
  for (int j = 0; j < 16; ++j) {
    const int k0 = frag_k(lane, j);
    a0[j] = (_Float16)in[rbase + k0];
    a1[j] = (_Float16)in[rbase + 32 + k0];
  }
  for (int g = 0; g < 4; ++g) {
    const int ncol = g * 16 + m;
    v16h b0, b1;
#pragma unroll
    for (int j = 0; j < 16; ++j) {
      const int k0 = frag_k(lane, j);
      b0[j] = s_b[k0 * H_DIM + ncol];
      b1[j] = s_b[(32 + k0) * H_DIM + ncol];
    }
    v8f c = {0,0,0,0,0,0,0,0};
    c = __builtin_amdgcn_wmma_f32_16x16x32_f16(false, a0, false, b0, (short)0, c, false, false);
    c = __builtin_amdgcn_wmma_f32_16x16x32_f16(false, a1, false, b1, (short)0, c, false, false);
#pragma unroll
    for (int r = 0; r < 8; ++r) {
      const int mm = r + hi * 8;
      const long long o = (long long)(row0 + mm) * H_DIM + ncol;
      if (chunk == 0) acc[o] = c[r] + filt_b[ncol];
      else            acc[o] += c[r];
    }
  }
}

// ---------------------------------------------------------------------------
// Decoder: out = z @ dec_w + dec_b, fused with b n (t f) -> b t n f scatter.
// N=24 padded to 32 (2 WMMA N-tiles).
// ---------------------------------------------------------------------------
__global__ __launch_bounds__(256)
void dec_kernel(const float* __restrict__ acc, const float* __restrict__ dec_w,
                const float* __restrict__ dec_b, float* __restrict__ out) {
  __shared__ _Float16 s_b[H_DIM * 32];   // 4 KB zero-padded
  const int tid  = threadIdx.x;
  const int lane = tid & 31;
  const int wv   = tid >> 5;
  const int m    = lane & 15;
  const int hi   = lane >> 4;
  for (int i = tid; i < H_DIM * 32; i += 256) {
    const int k = i >> 5, nn = i & 31;
    s_b[i] = (nn < 24) ? (_Float16)dec_w[k * 24 + nn] : (_Float16)0.0f;
  }
  __syncthreads();

  const int row0 = (blockIdx.x * 8 + wv) * 16;
  const long long rbase = (long long)(row0 + m) * H_DIM;
  v16h a0, a1;
#pragma unroll
  for (int j = 0; j < 16; ++j) {
    const int k0 = frag_k(lane, j);
    a0[j] = (_Float16)acc[rbase + k0];
    a1[j] = (_Float16)acc[rbase + 32 + k0];
  }
  for (int g = 0; g < 2; ++g) {
    const int ncol = g * 16 + m;
    v16h b0, b1;
#pragma unroll
    for (int j = 0; j < 16; ++j) {
      const int k0 = frag_k(lane, j);
      b0[j] = s_b[k0 * 32 + ncol];
      b1[j] = s_b[(32 + k0) * 32 + ncol];
    }
    v8f c = {0,0,0,0,0,0,0,0};
    c = __builtin_amdgcn_wmma_f32_16x16x32_f16(false, a0, false, b0, (short)0, c, false, false);
    c = __builtin_amdgcn_wmma_f32_16x16x32_f16(false, a1, false, b1, (short)0, c, false, false);
    if (ncol < 24) {                       // divergence only on store side
      const int th = ncol >> 1, f = ncol & 1;
      const float bias = dec_b[ncol];
#pragma unroll
      for (int r = 0; r < 8; ++r) {
        const int mm  = r + hi * 8;
        const int row = row0 + mm;
        const int bb  = row / N_DIM, nn = row % N_DIM;
        out[(((long long)bb * HOR + th) * N_DIM + nn) * F_INN + f] = c[r] + bias;
      }
    }
  }
}

// ---------------------------------------------------------------------------
extern "C" void kernel_launch(void* const* d_in, const int* in_sizes, int n_in,
                              void* d_out, int out_size, void* d_ws, size_t ws_size,
                              hipStream_t stream) {
  (void)in_sizes; (void)n_in; (void)out_size; (void)ws_size;
  const float* x        = (const float*)d_in[0];
  const int*   eidx     = (const int*)d_in[1];
  const float* ew       = (const float*)d_in[2];
  const float* enc_w    = (const float*)d_in[3];
  const float* enc_b    = (const float*)d_in[4];
  const float* node_emb = (const float*)d_in[5];
  const float* w_ih     = (const float*)d_in[6];
  const float* w_hh     = (const float*)d_in[7];
  const float* b_ih     = (const float*)d_in[8];
  const float* b_hh     = (const float*)d_in[9];
  const float* filt_w   = (const float*)d_in[10];
  const float* filt_b   = (const float*)d_in[11];
  const float* dec_w    = (const float*)d_in[12];
  const float* dec_b    = (const float*)d_in[13];
  float* out = (float*)d_out;

  const int* src = eidx;
  const int* tgt = eidx + E_DIM;

  // Workspace layout (floats): 4 x [BN,64] + 2 x [E] + 2 x [N]  (~167 MB)
  float* ws   = (float*)d_ws;
  size_t off  = 0;
  float* h    = ws + off; off += (size_t)BN * H_DIM;
  float* hopA = ws + off; off += (size_t)BN * H_DIM;
  float* hopB = ws + off; off += (size_t)BN * H_DIM;
  float* acc  = ws + off; off += (size_t)BN * H_DIM;
  float* wf   = ws + off; off += E_DIM;
  float* wb   = ws + off; off += E_DIM;
  float* degf = ws + off; off += N_DIM;
  float* degb = ws + off; off += N_DIM;   // degf/degb contiguous

  const size_t hopBytes = (size_t)BN * H_DIM * sizeof(float);
  const long long ptot  = (long long)E_DIM * B_DIM * H_DIM;
  const int pblocks     = (int)((ptot + 255) / 256);
  const int eblocks     = (E_DIM + 255) / 256;
  const int mblocks     = BN / 128;   // 1250

  // dynamic LDS for the GRU kernel: shared weights + per-wave state
  const size_t gruShmem =
      (size_t)(2 * G3 * H_DIM + GRU_WAVES * (16 * H_DIM + 2 * 16 * G3)) *
      sizeof(_Float16);   // 104 KB

  // 1) edge-weight normalization
  hipMemsetAsync(degf, 0, 2 * (size_t)N_DIM * sizeof(float), stream);
  deg_kernel<<<eblocks, 256, 0, stream>>>(ew, src, tgt, degf, degb);
  norm_kernel<<<eblocks, 256, 0, stream>>>(ew, src, tgt, degf, degb, wf, wb);

  // 2) encode + GRU -> h
  gru_encode_kernel<<<BN / (16 * GRU_WAVES), 32 * GRU_WAVES, gruShmem, stream>>>(
      x, enc_w, enc_b, node_emb, w_ih, w_hh, b_ih, b_hh, h);

  // 3) diffusion conv, accumulated chunk-by-chunk into acc
  filt_kernel<<<mblocks, 256, 0, stream>>>(h, filt_w, filt_b, acc, 0);

  hipMemsetAsync(hopA, 0, hopBytes, stream);
  prop_kernel<<<pblocks, 256, 0, stream>>>(h, src, tgt, wf, hopA);
  filt_kernel<<<mblocks, 256, 0, stream>>>(hopA, filt_w, filt_b, acc, 1);

  hipMemsetAsync(hopB, 0, hopBytes, stream);
  prop_kernel<<<pblocks, 256, 0, stream>>>(hopA, src, tgt, wf, hopB);
  filt_kernel<<<mblocks, 256, 0, stream>>>(hopB, filt_w, filt_b, acc, 2);

  hipMemsetAsync(hopA, 0, hopBytes, stream);
  prop_kernel<<<pblocks, 256, 0, stream>>>(h, tgt, src, wb, hopA);
  filt_kernel<<<mblocks, 256, 0, stream>>>(hopA, filt_w, filt_b, acc, 3);

  hipMemsetAsync(hopB, 0, hopBytes, stream);
  prop_kernel<<<pblocks, 256, 0, stream>>>(hopA, tgt, src, wb, hopB);
  filt_kernel<<<mblocks, 256, 0, stream>>>(hopB, filt_w, filt_b, acc, 4);

  // 4) decode + transpose
  dec_kernel<<<mblocks, 256, 0, stream>>>(acc, dec_w, dec_b, out);
}
